// NonLocalBlock_12343736008934
// MI455X (gfx1250) — compile-verified, matrix operand order checked by reading, not run on hardware
//
#include <hip/hip_runtime.h>
#include <hip/hip_bf16.h>
#include <stdint.h>

// ---------------------------------------------------------------------------
// NonLocalBlock (self-attention over 64x64 pixels) for MI455X / gfx1250.
// Pipeline: [threefry mask bits] + [f16 preconversion of x and weights]
//        -> [qkv projection (WMMA, all-b128 operands)]
//        -> [flash attention (WMMA, async_to_lds double-buffered K/V)]
//        -> [z projection + residual (WMMA)]
// ---------------------------------------------------------------------------

#define HW   4096   // h*w
#define CIN  256    // N channels
#define CH   128    // c = N/2
#define NB   4      // batch

typedef __attribute__((ext_vector_type(16))) _Float16 v16h;
typedef __attribute__((ext_vector_type(8)))  _Float16 v8h;
typedef __attribute__((ext_vector_type(8)))  float    v8f;

union V16u { v16h v; v8h h[2]; _Float16 e[16]; };
union V8u  { v8h h; _Float16 e[8]; };

__device__ __forceinline__ v8f wmma16(v16h a, v16h b, v8f c) {
  // D(16x16 f32) = A(16x32 f16) * B(32x16 f16) + C
  return __builtin_amdgcn_wmma_f32_16x16x32_f16(
      /*neg_a=*/false, a, /*neg_b=*/false, b,
      /*c_mod=*/(short)0, c, /*reuse_a=*/false, /*reuse_b=*/false);
}

// A/B fragment loader from f16 memory (global or LDS) where the WMMA
// K-dimension is the contiguous axis: two 16-byte loads per lane.
// A layout: lane l holds row m = l&15; halves 0..7 = K[kh..kh+7],
//           halves 8..15 = K[16+kh..16+kh+7], kh = (l>>4)*8.
// B is the mirrored column version (row param = column base).
__device__ __forceinline__ v16h frag_f16_contig(const _Float16* p, int row,
                                                int ld, int koff, int lane) {
  const int lo = lane & 15, kh = (lane >> 4) << 3;
  const _Float16* q = p + (size_t)(row + lo) * ld + koff + kh;
  V16u u;
  u.h[0] = *(const v8h*)(q);
  u.h[1] = *(const v8h*)(q + 16);
  return u.v;
}

// gfx1250 async global->LDS copy, 16 bytes per lane, tracked by ASYNCcnt.
__device__ __forceinline__ void async_ld16(uint32_t lds_off, uint64_t gaddr) {
  asm volatile("global_load_async_to_lds_b128 %0, %1, off"
               :: "v"(lds_off), "v"(gaddr)
               : "memory");
}
__device__ __forceinline__ void wait_async0() {
  asm volatile("s_wait_asynccnt 0" ::: "memory");
}

// ---------------------------------------------------------------------------
// Kernel 1: JAX threefry2x32 mask bits, key = (0, 42), prob 0.1.
// uniform() flattens to 2^26 counters split (i, i+2^25); out0 -> flat i,
// out1 -> flat i+2^25. Each thread handles 16 consecutive counters -> two
// aligned uint16 bitfield stores (no atomics).
// ---------------------------------------------------------------------------
__device__ __forceinline__ uint32_t rotl32(uint32_t x, int n) {
  return (x << n) | (x >> (32 - n));
}

__device__ __forceinline__ void threefry2x32_k42(uint32_t x0, uint32_t x1,
                                                 uint32_t& o0, uint32_t& o1) {
  const uint32_t ks0 = 0u, ks1 = 42u, ks2 = 0x1BD11BDAu ^ ks0 ^ ks1;
  const int ra[4] = {13, 15, 26, 6};
  const int rb[4] = {17, 29, 16, 24};
  x0 += ks0; x1 += ks1;
#pragma unroll
  for (int i = 0; i < 4; ++i) { x0 += x1; x1 = rotl32(x1, ra[i]); x1 ^= x0; }
  x0 += ks1; x1 += ks2 + 1u;
#pragma unroll
  for (int i = 0; i < 4; ++i) { x0 += x1; x1 = rotl32(x1, rb[i]); x1 ^= x0; }
  x0 += ks2; x1 += ks0 + 2u;
#pragma unroll
  for (int i = 0; i < 4; ++i) { x0 += x1; x1 = rotl32(x1, ra[i]); x1 ^= x0; }
  x0 += ks0; x1 += ks1 + 3u;
#pragma unroll
  for (int i = 0; i < 4; ++i) { x0 += x1; x1 = rotl32(x1, rb[i]); x1 ^= x0; }
  x0 += ks1; x1 += ks2 + 4u;
#pragma unroll
  for (int i = 0; i < 4; ++i) { x0 += x1; x1 = rotl32(x1, ra[i]); x1 ^= x0; }
  x0 += ks2; x1 += ks0 + 5u;
  o0 = x0; o1 = x1;
}

__device__ __forceinline__ uint32_t bits_lt_p(uint32_t bits) {
  float u = __uint_as_float((bits >> 9) | 0x3f800000u) - 1.0f;
  return u < 0.1f ? 1u : 0u;
}

__global__ __launch_bounds__(256) void mask_kernel(uint16_t* __restrict__ mask16) {
  const uint32_t HALF = 1u << 25;
  const uint32_t tid = blockIdx.x * 256u + threadIdx.x;  // 0 .. 2^21-1
  const uint32_t base = tid << 4;
  uint32_t a = 0u, b = 0u;
#pragma unroll 4
  for (uint32_t j = 0; j < 16; ++j) {
    uint32_t o0, o1;
    threefry2x32_k42(base + j, base + j + HALF, o0, o1);
    a |= bits_lt_p(o0) << j;
    b |= bits_lt_p(o1) << j;
  }
  mask16[tid] = (uint16_t)a;
  mask16[tid + (HALF >> 4)] = (uint16_t)b;
}

// ---------------------------------------------------------------------------
// Kernel 2a: transpose-convert x[nb][256][4096] f32 -> xh[nb][4096][256] f16
// (LDS 32x33 tile; both global access patterns coalesced).
// ---------------------------------------------------------------------------
__global__ __launch_bounds__(256) void xpose_kernel(const float* __restrict__ x,
                                                    _Float16* __restrict__ xh) {
  __shared__ float tile[32][33];
  const int tx = threadIdx.x, ty = threadIdx.y;
  const int t0 = blockIdx.x * 32, c0 = blockIdx.y * 32, nb = blockIdx.z;
  const float* xn = x + (size_t)nb * CIN * HW;
#pragma unroll
  for (int i = ty; i < 32; i += 8)
    tile[i][tx] = xn[(size_t)(c0 + i) * HW + t0 + tx];
  __syncthreads();
  _Float16* xo = xh + (size_t)nb * HW * CIN;
#pragma unroll
  for (int i = ty; i < 32; i += 8)
    xo[(size_t)(t0 + i) * CIN + c0 + tx] = (_Float16)tile[tx][i];
}

// Kernel 2b: convert wq/wk/wv/wz (f32) to one packed f16 buffer.
__global__ __launch_bounds__(256) void wconv_kernel(
    const float* __restrict__ wq, const float* __restrict__ wk,
    const float* __restrict__ wv, const float* __restrict__ wz,
    _Float16* __restrict__ dst) {
  const int SZ = CH * CIN;  // 32768 per matrix
  int i = blockIdx.x * 256 + threadIdx.x;  // 0 .. 4*SZ-1
  int sel = i / SZ, r = i - sel * SZ;
  const float* s = (sel == 0) ? wq : (sel == 1) ? wk : (sel == 2) ? wv : wz;
  dst[i] = (_Float16)s[r];
}

// ---------------------------------------------------------------------------
// Kernel 3: Q/K/V projection via WMMA; all operands are f16/b128 loads now.
// One wave computes a 16(out-ch) x 16(pixel) tile, k-loop over 256 channels.
// Q,K stored [t][c] f16; V stored transposed [c][s] f16.
// ---------------------------------------------------------------------------
__global__ __launch_bounds__(32) void qkv_proj_kernel(
    const _Float16* __restrict__ xh, const _Float16* __restrict__ wh,
    const float* __restrict__ bq, const float* __restrict__ bk,
    const float* __restrict__ bv, _Float16* __restrict__ Q,
    _Float16* __restrict__ K, _Float16* __restrict__ Vt) {
  const int lane = threadIdx.x & 31;
  const int t0 = blockIdx.x * 16;    // pixel tile
  const int c0 = blockIdx.y * 16;    // output-channel tile
  const int which = blockIdx.z % 3;  // 0=q 1=k 2=v
  const int nb = blockIdx.z / 3;
  const _Float16* w = wh + (size_t)which * CH * CIN;  // [128][256] f16
  const float* bias = (which == 0) ? bq : (which == 1) ? bk : bv;
  const _Float16* xn = xh + (size_t)nb * HW * CIN;    // [t][ch] f16

  v8f acc = {};
#pragma unroll
  for (int kc = 0; kc < CIN; kc += 32) {
    v16h a = frag_f16_contig(w, c0, CIN, kc, lane);   // w[c0+m][kc+kk]
    v16h b = frag_f16_contig(xn, t0, CIN, kc, lane);  // x[t0+n][kc+kk]
    acc = wmma16(a, b, acc);
  }

  const int lo = lane & 15, hi = lane >> 4;
  if (which < 2) {
    _Float16* dst = ((which == 0) ? Q : K) +
                    ((size_t)nb * HW + t0 + lo) * CH + c0 + hi * 8;
    V8u o;
#pragma unroll
    for (int r = 0; r < 8; ++r)
      o.e[r] = (_Float16)(acc[r] + bias[c0 + hi * 8 + r]);
    *(v8h*)dst = o.h;
  } else {
#pragma unroll
    for (int r = 0; r < 8; ++r) {
      int c = c0 + hi * 8 + r;
      Vt[((size_t)nb * CH + c) * HW + t0 + lo] = (_Float16)(acc[r] + bias[c]);
    }
  }
}

// ---------------------------------------------------------------------------
// Kernel 4: fused flash attention. 4 waves/block, each owning 16 query rows;
// the block cooperatively double-buffers each 32-column K chunk (8 KB) and
// V chunk (8 KB) into LDS with global_load_async_to_lds_b128 (ASYNCcnt),
// then every wave feeds its WMMAs from ds_load_b128. The 4096x4096 logits
// are never materialized. Online softmax via shfl_xor row reductions;
// P-tile (C layout) -> A layout transpose through LDS.
// ---------------------------------------------------------------------------
#define ATTN_WAVES 4
__global__ __launch_bounds__(32 * ATTN_WAVES) void attn_kernel(
    const _Float16* __restrict__ Q, const _Float16* __restrict__ K,
    const _Float16* __restrict__ Vt, const uint32_t* __restrict__ maskw,
    _Float16* __restrict__ Z) {
  __shared__ __align__(16) _Float16 Kb[2][32][CH];          // 16 KB
  __shared__ __align__(16) _Float16 Vb[2][CH][32];          // 16 KB
  __shared__ __align__(16) _Float16 Pl[ATTN_WAVES][16][32]; // 4 KB
  const int lane = threadIdx.x & 31;
  const int wid = threadIdx.x >> 5;
  const int nb = blockIdx.y;
  const int t0 = (blockIdx.x * ATTN_WAVES + wid) * 16;
  const int lo = lane & 15, hi = lane >> 4;
  const _Float16* Qn = Q + (size_t)nb * HW * CH;
  const _Float16* Kn = K + (size_t)nb * HW * CH;
  const _Float16* Vn = Vt + (size_t)nb * CH * HW;

  // stage one 32-column chunk of K and V into LDS buffer nbuf (async)
  auto issue_chunk = [&](int s0c, int nbuf) {
    // K chunk: 32 rows x 256 B; wave wid moves rows wid*8..wid*8+7
#pragma unroll
    for (int t = 0; t < 4; ++t) {
      int row = wid * 8 + t * 2 + (lane >> 4);
      uint32_t boff = (uint32_t)(lane & 15) * 16u;
      uint64_t g = (uint64_t)(uintptr_t)(Kn + (size_t)(s0c + row) * CH) + boff;
      uint32_t l = (uint32_t)(uintptr_t)(&Kb[nbuf][row][0]) + boff;
      async_ld16(l, g);
    }
    // V chunk: 128 rows x 64 B; wave wid moves rows wid*32..wid*32+31
#pragma unroll
    for (int t = 0; t < 4; ++t) {
      int c = wid * 32 + t * 8 + (lane >> 2);
      uint32_t boff = (uint32_t)(lane & 3) * 16u;
      uint64_t g = (uint64_t)(uintptr_t)(Vn + (size_t)c * HW + s0c) + boff;
      uint32_t l = (uint32_t)(uintptr_t)(&Vb[nbuf][c][0]) + boff;
      async_ld16(l, g);
    }
  };

  v16h aQ[4];
#pragma unroll
  for (int kc = 0; kc < 4; ++kc)
    aQ[kc] = frag_f16_contig(Qn, t0, CH, kc * 32, lane);

  const v8f vzero = {};
  v8f acc[8];
#pragma unroll
  for (int j = 0; j < 8; ++j) acc[j] = vzero;
  float rmax[8], rsum[8];
#pragma unroll
  for (int r = 0; r < 8; ++r) { rmax[r] = -3.0e38f; rsum[r] = 0.0f; }

  const float scale = 1.0f / 64.0f;  // 1/sqrt(hw)
  const size_t mrow = (size_t)(nb * HW + t0 + hi * 8) * (HW / 32);

  issue_chunk(0, 0);
  for (int s0 = 0; s0 < HW; s0 += 32) {
    const int buf = (s0 >> 5) & 1;
    wait_async0();     // own async stores into Kb/Vb[buf] complete
    __syncthreads();   // whole chunk visible; previous reads of buf^1 done
    if (s0 + 32 < HW) issue_chunk(s0 + 32, buf ^ 1);  // overlap with compute

    // S = Q * K^T for 32 columns (two 16x16 tiles), k=128 over 4 WMMAs each
    const _Float16* Kl = &Kb[buf][0][0];
    v8f S0 = vzero, S1 = vzero;
#pragma unroll
    for (int kc = 0; kc < 4; ++kc) {
      v16h b0 = frag_f16_contig(Kl, 0, CH, kc * 32, lane);
      v16h b1 = frag_f16_contig(Kl, 16, CH, kc * 32, lane);
      S0 = wmma16(aQ[kc], b0, S0);
      S1 = wmma16(aQ[kc], b1, S1);
    }

    // scale + precomputed threefry mask (one u32 = 32 columns per row)
    uint32_t mb[8];
#pragma unroll
    for (int r = 0; r < 8; ++r)
      mb[r] = maskw[mrow + (size_t)r * (HW / 32) + (s0 >> 5)];
#pragma unroll
    for (int r = 0; r < 8; ++r) {
      float v0 = S0[r] * scale, v1 = S1[r] * scale;
      S0[r] = ((mb[r] >> lo) & 1u) ? -1.0e9f : v0;
      S1[r] = ((mb[r] >> (16 + lo)) & 1u) ? -1.0e9f : v1;
    }

    // online softmax: row stats reduced across the 16 lanes sharing rows
    float cm[8];
#pragma unroll
    for (int r = 0; r < 8; ++r) cm[r] = fmaxf(S0[r], S1[r]);
    for (int m = 1; m < 16; m <<= 1) {
#pragma unroll
      for (int r = 0; r < 8; ++r)
        cm[r] = fmaxf(cm[r], __shfl_xor(cm[r], m, 32));
    }
    float nmax[8], corr[8], psum[8];
    v8f P0, P1;
#pragma unroll
    for (int r = 0; r < 8; ++r) {
      nmax[r] = fmaxf(rmax[r], cm[r]);
      corr[r] = __expf(rmax[r] - nmax[r]);
      P0[r] = __expf(S0[r] - nmax[r]);
      P1[r] = __expf(S1[r] - nmax[r]);
      psum[r] = P0[r] + P1[r];
    }
    for (int m = 1; m < 16; m <<= 1) {
#pragma unroll
      for (int r = 0; r < 8; ++r) psum[r] += __shfl_xor(psum[r], m, 32);
    }
#pragma unroll
    for (int r = 0; r < 8; ++r) {
      rsum[r] = rsum[r] * corr[r] + psum[r];
      rmax[r] = nmax[r];
    }
#pragma unroll
    for (int j = 0; j < 8; ++j)
#pragma unroll
      for (int r = 0; r < 8; ++r) acc[j][r] *= corr[r];

    // transpose P (C layout) into an A fragment through LDS
#pragma unroll
    for (int r = 0; r < 8; ++r) {
      Pl[wid][hi * 8 + r][lo] = (_Float16)P0[r];
      Pl[wid][hi * 8 + r][16 + lo] = (_Float16)P1[r];
    }
    __syncthreads();
    v16h aP = frag_f16_contig(&Pl[wid][0][0], 0, 32, 0, lane);

    // O += P * V (8 column tiles of 16 channels, k=32) from LDS
    const _Float16* Vl = &Vb[buf][0][0];
#pragma unroll
    for (int j = 0; j < 8; ++j) {
      v16h bV = frag_f16_contig(Vl, j * 16, 32, 0, lane);
      acc[j] = wmma16(aP, bV, acc[j]);
    }
  }

  // normalize and store Z[t][c] f16
  float rinv[8];
#pragma unroll
  for (int r = 0; r < 8; ++r) rinv[r] = 1.0f / rsum[r];
#pragma unroll
  for (int j = 0; j < 8; ++j)
#pragma unroll
    for (int r = 0; r < 8; ++r)
      Z[((size_t)nb * HW + t0 + hi * 8 + r) * CH + j * 16 + lo] =
          (_Float16)(acc[j][r] * rinv[r]);
}

// ---------------------------------------------------------------------------
// Kernel 5: output projection wz[256x128] @ Z + bz + residual x.
// ---------------------------------------------------------------------------
__global__ __launch_bounds__(32) void zproj_kernel(
    const float* __restrict__ x, const _Float16* __restrict__ wzh,
    const float* __restrict__ bz, const _Float16* __restrict__ Z,
    float* __restrict__ out) {
  const int lane = threadIdx.x & 31;
  const int t0 = blockIdx.x * 16;
  const int o0 = blockIdx.y * 16;
  const int nb = blockIdx.z;
  const _Float16* Zn = Z + (size_t)nb * HW * CH;

  v8f acc = {};
#pragma unroll
  for (int kc = 0; kc < CH; kc += 32) {
    v16h a = frag_f16_contig(wzh, o0, CH, kc, lane);  // wz[o0+m][kc+kk]
    v16h b = frag_f16_contig(Zn, t0, CH, kc, lane);   // Z[t0+n][kc+kk]
    acc = wmma16(a, b, acc);
  }

  const int lo = lane & 15, hi = lane >> 4;
#pragma unroll
  for (int r = 0; r < 8; ++r) {
    int o = o0 + hi * 8 + r;
    size_t idx = ((size_t)nb * CIN + o) * HW + t0 + lo;
    out[idx] = x[idx] + acc[r] + bz[o];
  }
}

// ---------------------------------------------------------------------------
extern "C" void kernel_launch(void* const* d_in, const int* in_sizes, int n_in,
                              void* d_out, int out_size, void* d_ws,
                              size_t ws_size, hipStream_t stream) {
  const float* x  = (const float*)d_in[0];
  const float* wq = (const float*)d_in[1];
  const float* bq = (const float*)d_in[2];
  const float* wk = (const float*)d_in[3];
  const float* bk = (const float*)d_in[4];
  const float* wv = (const float*)d_in[5];
  const float* bv = (const float*)d_in[6];
  const float* wz = (const float*)d_in[7];
  const float* bz = (const float*)d_in[8];
  float* out = (float*)d_out;

  // workspace layout (32 MiB + 256 KiB)
  char* ws = (char*)d_ws;
  _Float16* Q      = (_Float16*)(ws + (size_t)0);            // 4 MiB
  _Float16* K      = (_Float16*)(ws + ((size_t)4 << 20));    // 4 MiB
  _Float16* Vt     = (_Float16*)(ws + ((size_t)8 << 20));    // 4 MiB
  _Float16* Z      = (_Float16*)(ws + ((size_t)12 << 20));   // 4 MiB
  uint16_t* mask16 = (uint16_t*)(ws + ((size_t)16 << 20));   // 8 MiB
  _Float16* xh     = (_Float16*)(ws + ((size_t)24 << 20));   // 8 MiB
  _Float16* wh     = (_Float16*)(ws + ((size_t)32 << 20));   // 256 KiB: q,k,v,z
  _Float16* wzh    = wh + (size_t)3 * CH * CIN;

  mask_kernel<<<dim3((1u << 21) / 256), 256, 0, stream>>>(mask16);
  wconv_kernel<<<dim3(4 * CH * CIN / 256), 256, 0, stream>>>(wq, wk, wv, wz, wh);
  xpose_kernel<<<dim3(HW / 32, CIN / 32, NB), dim3(32, 8), 0, stream>>>(x, xh);
  qkv_proj_kernel<<<dim3(HW / 16, CH / 16, 3 * NB), 32, 0, stream>>>(
      xh, wh, bq, bk, bv, Q, K, Vt);
  attn_kernel<<<dim3(HW / 16 / ATTN_WAVES, NB), 32 * ATTN_WAVES, 0, stream>>>(
      Q, K, Vt, (const uint32_t*)mask16, Z);
  zproj_kernel<<<dim3(HW / 16, CIN / 16, NB), 32, 0, stream>>>(x, wzh, bz, Z,
                                                               out);
}
